// CustomBertAttention_38663295598861
// MI455X (gfx1250) — matrix operand order repaired and checked by reading, not compiled, not run on hardware
//
#include <hip/hip_runtime.h>

typedef _Float16 f16;
typedef __attribute__((ext_vector_type(4)))  _Float16 v4h;
typedef __attribute__((ext_vector_type(8)))  _Float16 v8h;
typedef __attribute__((ext_vector_type(16))) _Float16 v16h;
typedef __attribute__((ext_vector_type(8)))  float    v8f;
typedef __attribute__((ext_vector_type(4)))  int      v4i;

constexpr int HIDDEN   = 1024;
constexpr int HEADS    = 16;
constexpr int HEAD_DIM = 64;
constexpr int BATCH    = 2;
constexpr int SEQ      = 2048;
constexpr int NROWS    = BATCH * SEQ;  // 4096

// LDS row strides (f16 units): multiples of 8 f16 (16B -> b128-aligned) with
// conflict-free bank strides (80B -> 20 banks, 144B -> 36 banks mod 64).
constexpr int LDA = 40;  // for 32-wide tiles (+8 pad)
constexpr int LDK = 72;  // for 64-wide tiles (+8 pad)

#if __has_builtin(__builtin_amdgcn_global_load_async_to_lds_b128)
#define HAVE_ASYNC_LDS 1
typedef __attribute__((address_space(1))) v4i* gptr_v4i;
typedef __attribute__((address_space(3))) v4i* lptr_v4i;
#endif

__device__ __forceinline__ void async_copy_b128(const void* gsrc, void* ldst) {
#ifdef HAVE_ASYNC_LDS
    __builtin_amdgcn_global_load_async_to_lds_b128(
        (gptr_v4i)(void*)gsrc, (lptr_v4i)ldst, /*offset=*/0, /*cpol=*/0);
#else
    *(v8h*)ldst = *(const v8h*)gsrc;
#endif
}

__device__ __forceinline__ void wait_async_then_barrier() {
#ifdef HAVE_ASYNC_LDS
#if __has_builtin(__builtin_amdgcn_s_wait_asynccnt)
    __builtin_amdgcn_s_wait_asynccnt(0);
#else
    asm volatile("s_wait_asynccnt 0x0" ::: "memory");
#endif
#endif
    __syncthreads();
}

__device__ __forceinline__ v16h cat16(v8h lo, v8h hi) {
    return __builtin_shufflevector(lo, hi, 0,1,2,3,4,5,6,7,8,9,10,11,12,13,14,15);
}

__device__ __forceinline__ v8f wmma_f16(v16h a, v16h b, v8f c) {
    return __builtin_amdgcn_wmma_f32_16x16x32_f16(
        false, a, false, b, (short)0, c, false, false);
}

// A-fragment (16x32 f16) from row-major LDS tile with row stride LD.
// lanes 0-15: M=lane, K = 0..7 | 16..23 ; lanes 16-31: M=lane-16, K = 8..15 | 24..31
template<int LD>
__device__ __forceinline__ v16h a_frag(const f16* base, int lane) {
    const int m  = lane & 15;
    const int k0 = (lane & 16) ? 8 : 0;
    const f16* row = base + m * LD + k0;
    return cat16(*(const v8h*)(row), *(const v8h*)(row + 16));
}

// B-fragment (32x16 f16) where element (k,n) = T[n*LD + k] (tile stored N-major).
// lane&15 = N ; lanes 0-15: K = 0..7 | 16..23 ; lanes 16-31: K = 8..15 | 24..31.
template<int LD>
__device__ __forceinline__ v16h b_fragT(const f16* baseT, int lane) {
    const int n  = lane & 15;
    const int k0 = (lane & 16) ? 8 : 0;
    const f16* p = baseT + n * LD + k0;
    return cat16(*(const v8h*)(p), *(const v8h*)(p + 16));
}

// ---------------------------------------------------------------------------
// Kernel 1: QKV projection.  Y = X*W + b -> f16 workspace.
// grid = (HIDDEN/64, NROWS/128, 3) ; block = 128 (4 waves).
// Each wave: 32 rows x 64 cols = 8 WMMA accumulators; K loop step 32.
// ---------------------------------------------------------------------------
__global__ __launch_bounds__(128) void qkv_proj_kernel(
    const float* __restrict__ X,
    const float* __restrict__ Wq, const float* __restrict__ bq,
    const float* __restrict__ Wk, const float* __restrict__ bk,
    const float* __restrict__ Wv, const float* __restrict__ bv,
    f16* __restrict__ qkv)
{
    const float* W;
    const float* bias;
    f16* Y;
    const int z = blockIdx.z;
    if (z == 0)      { W = Wq; bias = bq; Y = qkv; }
    else if (z == 1) { W = Wk; bias = bk; Y = qkv + (size_t)NROWS * HIDDEN; }
    else             { W = Wv; bias = bv; Y = qkv + (size_t)2 * NROWS * HIDDEN; }

    __shared__ f16 Xs [128 * LDA];  // X tile, row-major [128][32] (+pad)
    __shared__ f16 WsT[64 * LDA];   // W tile, stored transposed [n=64][k=32] (+pad)

    const int tid  = threadIdx.x;
    const int lane = tid & 31;
    const int wave = tid >> 5;
    const int row0 = blockIdx.y * 128;
    const int col0 = blockIdx.x * 64;

    v8f acc[2][4] = {};

    for (int k0 = 0; k0 < HIDDEN; k0 += 32) {
        // Prefetch next K-step tiles into cache (global_prefetch_b8).
        if (k0 + 32 < HIDDEN) {
            __builtin_prefetch(X + (size_t)(row0 + (tid >> 3)) * HIDDEN + k0 + 32, 0, 1);
            __builtin_prefetch(W + (size_t)(k0 + 32 + (tid >> 4)) * HIDDEN + col0, 0, 1);
        }
        // Stage X tile: 128x32 f32 -> f16.  1024 float4 chunks, 8 per thread.
#pragma unroll
        for (int j = 0; j < 8; ++j) {
            const int cid = tid + 128 * j;
            const int r = cid >> 3, c = (cid & 7) * 4;
            const float4 xv = *(const float4*)(X + (size_t)(row0 + r) * HIDDEN + k0 + c);
            v4h hx = { (f16)xv.x, (f16)xv.y, (f16)xv.z, (f16)xv.w };
            *(v4h*)&Xs[r * LDA + c] = hx;
        }
        // Stage W tile transposed: 32x64 f32 -> WsT[n][k].  512 chunks, 4/thread.
#pragma unroll
        for (int j = 0; j < 4; ++j) {
            const int cid = tid + 128 * j;
            const int r = cid >> 4, c = (cid & 15) * 4;   // r=k-local, c=n-local
            const float4 wv = *(const float4*)(W + (size_t)(k0 + r) * HIDDEN + col0 + c);
            WsT[(c + 0) * LDA + r] = (f16)wv.x;
            WsT[(c + 1) * LDA + r] = (f16)wv.y;
            WsT[(c + 2) * LDA + r] = (f16)wv.z;
            WsT[(c + 3) * LDA + r] = (f16)wv.w;
        }
        __syncthreads();

        const v16h a0 = a_frag<LDA>(&Xs[(wave * 32) * LDA], lane);
        const v16h a1 = a_frag<LDA>(&Xs[(wave * 32 + 16) * LDA], lane);
#pragma unroll
        for (int nt = 0; nt < 4; ++nt) {
            const v16h b = b_fragT<LDA>(&WsT[(nt * 16) * LDA], lane);
            acc[0][nt] = wmma_f16(a0, b, acc[0][nt]);
            acc[1][nt] = wmma_f16(a1, b, acc[1][nt]);
        }
        __syncthreads();
    }

    // Epilogue: add bias, store f16.  C layout: VGPR r -> row r+8*(lane>=16), col lane&15.
    const int n_lo = lane & 15;
    const int hi   = (lane & 16) ? 8 : 0;
#pragma unroll
    for (int mt = 0; mt < 2; ++mt) {
#pragma unroll
        for (int nt = 0; nt < 4; ++nt) {
            const int n = col0 + nt * 16 + n_lo;
            const float bb = bias[n];
#pragma unroll
            for (int r = 0; r < 8; ++r) {
                const int mrow = row0 + wave * 32 + mt * 16 + r + hi;
                Y[(size_t)mrow * HIDDEN + n] = (f16)(acc[mt][nt][r] + bb);
            }
        }
    }
}

// ---------------------------------------------------------------------------
// Kernel 2: fused per-head flash attention.
// grid = (SEQ/64, HEADS, BATCH) ; block = 128 (4 waves, each owns 16 q-rows).
// K tile staged via GLOBAL_LOAD_ASYNC_TO_LDS_B128 (ASYNCcnt) when available.
// ---------------------------------------------------------------------------
__global__ __launch_bounds__(128) void attn_kernel(
    const f16* __restrict__ qkv, float* __restrict__ out)
{
    const f16* Q = qkv;
    const f16* K = qkv + (size_t)NROWS * HIDDEN;
    const f16* V = qkv + (size_t)2 * NROWS * HIDDEN;

    __shared__ f16 Kt [32 * LDK];       // K tile row-major [s=32][d=64] (+pad)
    __shared__ f16 VtT[64 * LDA];       // V tile transposed [d=64][s=32] (+pad)
    __shared__ f16 Ps [4 * 16 * LDA];   // per-wave P tile [16][32] (+pad)

    const int tid  = threadIdx.x;
    const int lane = tid & 31;
    const int wave = tid >> 5;
    const int b  = blockIdx.z;
    const int h  = blockIdx.y;
    const int q0 = blockIdx.x * 64 + wave * 16;

    const int n_lo = lane & 15;
    const int hi8  = (lane & 16) ? 8 : 0;

    // Preload Q A-fragments (d 0..31, 32..63); per-lane chunks are 16B aligned.
    const f16* qrow = Q + (size_t)(b * SEQ + q0 + n_lo) * HIDDEN + h * HEAD_DIM + hi8;
    v16h aq[2];
    aq[0] = cat16(*(const v8h*)(qrow),      *(const v8h*)(qrow + 16));
    aq[1] = cat16(*(const v8h*)(qrow + 32), *(const v8h*)(qrow + 48));

    float mrow[8], lrow[8];
#pragma unroll
    for (int r = 0; r < 8; ++r) { mrow[r] = -1e30f; lrow[r] = 0.0f; }
    v8f o[4] = {};

    const float scale = 0.125f;  // 1/sqrt(64)
    f16* myP = &Ps[wave * 16 * LDA];

    for (int kb = 0; kb < SEQ; kb += 32) {
        __syncthreads();
        // Stage K (async copy to LDS, row-major) and V (transposed, synchronous).
        // 256 b128 chunks each, 2 per thread.
#pragma unroll
        for (int j = 0; j < 2; ++j) {
            const int cid = tid + 128 * j;
            const int r = cid >> 3, c = (cid & 7) * 8;   // r = s-local, c = d-local
            const size_t g = (size_t)(b * SEQ + kb + r) * HIDDEN + h * HEAD_DIM + c;
            async_copy_b128(K + g, &Kt[r * LDK + c]);
            const v8h vv = *(const v8h*)(V + g);
#pragma unroll
            for (int e = 0; e < 8; ++e) VtT[(c + e) * LDA + r] = vv[e];
            // Prefetch next tile's lines (global_prefetch_b8).
            if (kb + 32 < SEQ) {
                __builtin_prefetch(K + g + 32 * HIDDEN, 0, 1);
                __builtin_prefetch(V + g + 32 * HIDDEN, 0, 1);
            }
        }
        wait_async_then_barrier();

        // Scores: two 16x16 f32 tiles, contraction over d=64 (2 WMMA each).
        // Q.K^T B-frag: element (k=d, n=s) = Kt[s][d] -> b_fragT on row-major Kt.
        v8f c0 = {}, c1 = {};
        c0 = wmma_f16(aq[0], b_fragT<LDK>(&Kt[0],             lane), c0);
        c0 = wmma_f16(aq[1], b_fragT<LDK>(&Kt[32],            lane), c0);
        c1 = wmma_f16(aq[0], b_fragT<LDK>(&Kt[16 * LDK],      lane), c1);
        c1 = wmma_f16(aq[1], b_fragT<LDK>(&Kt[16 * LDK + 32], lane), c1);

        // Online softmax; rows live across 16-lane halves -> butterfly reduce.
#pragma unroll
        for (int r = 0; r < 8; ++r) {
            const float s0 = c0[r] * scale;
            const float s1 = c1[r] * scale;
            float tmax = fmaxf(s0, s1);
            tmax = fmaxf(tmax, __shfl_xor(tmax, 1, 32));
            tmax = fmaxf(tmax, __shfl_xor(tmax, 2, 32));
            tmax = fmaxf(tmax, __shfl_xor(tmax, 4, 32));
            tmax = fmaxf(tmax, __shfl_xor(tmax, 8, 32));

            const float newm  = fmaxf(mrow[r], tmax);
            const float alpha = __expf(mrow[r] - newm);
            const float p0 = __expf(s0 - newm);
            const float p1 = __expf(s1 - newm);

            float ps = p0 + p1;
            ps += __shfl_xor(ps, 1, 32);
            ps += __shfl_xor(ps, 2, 32);
            ps += __shfl_xor(ps, 4, 32);
            ps += __shfl_xor(ps, 8, 32);

            lrow[r] = lrow[r] * alpha + ps;
            mrow[r] = newm;
#pragma unroll
            for (int nt = 0; nt < 4; ++nt) o[nt][r] *= alpha;

            myP[(r + hi8) * LDA + n_lo]      = (f16)p0;
            myP[(r + hi8) * LDA + 16 + n_lo] = (f16)p1;
        }
        __syncthreads();

        // O += P(16x32) * V(32x64); B-frag element (k=s, n=d) = VtT[d][s].
        const v16h ap = a_frag<LDA>(myP, lane);
#pragma unroll
        for (int nt = 0; nt < 4; ++nt) {
            const v16h bvf = b_fragT<LDA>(&VtT[(nt * 16) * LDA], lane);
            o[nt] = wmma_f16(ap, bvf, o[nt]);
        }
    }

    // Epilogue: normalize and write fp32 output [B,S,HIDDEN].
#pragma unroll
    for (int r = 0; r < 8; ++r) {
        const int srow = q0 + r + hi8;
        const float invl = 1.0f / lrow[r];
#pragma unroll
        for (int nt = 0; nt < 4; ++nt) {
            out[(size_t)(b * SEQ + srow) * HIDDEN + h * HEAD_DIM + nt * 16 + n_lo] =
                o[nt][r] * invl;
        }
    }
}

extern "C" void kernel_launch(void* const* d_in, const int* in_sizes, int n_in,
                              void* d_out, int out_size, void* d_ws, size_t ws_size,
                              hipStream_t stream) {
    const float* X  = (const float*)d_in[0];
    const float* Wq = (const float*)d_in[1];
    const float* bq = (const float*)d_in[2];
    const float* Wk = (const float*)d_in[3];
    const float* bk = (const float*)d_in[4];
    const float* Wv = (const float*)d_in[5];
    const float* bv = (const float*)d_in[6];
    float* out = (float*)d_out;
    f16* qkv = (f16*)d_ws;  // Q | K | V, each NROWS*HIDDEN f16 (24 MB total)

    dim3 g1(HIDDEN / 64, NROWS / 128, 3);  // (16, 32, 3)
    qkv_proj_kernel<<<g1, 128, 0, stream>>>(X, Wq, bq, Wk, bk, Wv, bv, qkv);

    dim3 g2(SEQ / 64, HEADS, BATCH);       // (32, 16, 2)
    attn_kernel<<<g2, 128, 0, stream>>>(qkv, out);
}
